// LunarisMind_7241314861342
// MI455X (gfx1250) — matrix operand order repaired and checked by reading, not compiled, not run on hardware
//
#include <hip/hip_runtime.h>
#include <math.h>

// ---------------------------------------------------------------------------
// Model constants (match reference)
// ---------------------------------------------------------------------------
constexpr int Vv = 32000;
constexpr int Dm = 1024;
constexpr int Hh = 16;
constexpr int Ll = 8;
constexpr int Rr = 32;
constexpr int FFm = 4096;
constexpr int Bb = 2;
constexpr int Ss = 1024;
constexpr int HD = Dm / Hh;        // 64
constexpr int Mrows = Bb * Ss;     // 2048 tokens

// ---------------------------------------------------------------------------
// WMMA types (gfx1250, wave32)
// ---------------------------------------------------------------------------
typedef __attribute__((ext_vector_type(16))) __bf16 v16bf;
typedef __attribute__((ext_vector_type(8)))  float  v8f;

// Native bf16 conversion (RNE per C semantics) -> backend selects v_cvt_*_bf16
__device__ __forceinline__ unsigned short f2bf(float f) {
  return __builtin_bit_cast(unsigned short, (__bf16)f);
}
__device__ __forceinline__ unsigned pack2bf(float x, float y) {
  return (unsigned)f2bf(x) | ((unsigned)f2bf(y) << 16);
}

// ---------------------------------------------------------------------------
// Tiled WMMA GEMM: Out[M,N] = epilogue( alpha * (A[M,K] @ W[K,N]) + Cadd )
//   TRANSB: W stored [N,K] (tied LM head: tok_emb^T)
//   GUARDN: bounds-check N (only the rank-32 LoRA down-projections need it)
//   epilogue 0: Out = v                (v = alpha*acc + Cadd)
//   epilogue 1: Out = Xin + ls[n] * v  (LayerScale residual; ldc == D)
// Requires: M % 128 == 0, K % 32 == 0 (true for every call here).
// Block: 256 threads = 8 waves; block tile 128x128, K-step 32.
// Wave w: rows 32*(w&3) (2 row-frags), cols 64*(w>>2) (4 col-frags)
//   -> 8 WMMA accumulators per wave, 8 v_wmma per K-step.
// Double-buffered LDS: next tile's global loads issue before this tile's
// WMMAs; pack+store into the other buffer; ONE barrier per K-step.
// ---------------------------------------------------------------------------
#define BM 128
#define BN 128
#define BK 32
#define AS_STRIDE 48   // ushort stride (96 B): 16B/32B-aligned fragment rows
#define BS_STRIDE 48

template<bool GUARDN, bool TRANSB>
__global__ __launch_bounds__(256)
void gemm_wmma_bf16(const float* __restrict__ A, const float* __restrict__ W,
                    const float* __restrict__ Cadd, const float* __restrict__ Xin,
                    const float* __restrict__ ls, float* __restrict__ Out,
                    int Ndim, int Kdim, int lda, int ldw, int ldc,
                    float alpha, int epilogue)
{
  __shared__ unsigned short As[2][BM * AS_STRIDE];   // A tile, row-major [m][k]
  __shared__ unsigned short Bs[2][BN * BS_STRIDE];   // W tile, n-major  [n][k]

  const int tid  = threadIdx.x;
  const int lane = tid & 31;
  const int wave = tid >> 5;
  const int m0 = blockIdx.y * BM;
  const int n0 = blockIdx.x * BN;
  const int wr = (wave & 3) * 32;       // wave row base (2 x 16)
  const int wc = (wave >> 2) * 64;      // wave col base (4 x 16)
  const int lr = lane & 15;
  const int hi = lane >> 4;

  v8f acc[2][4];
#pragma unroll
  for (int rg = 0; rg < 2; ++rg)
#pragma unroll
    for (int g = 0; g < 4; ++g)
      acc[rg][g] = (v8f){0.f,0.f,0.f,0.f,0.f,0.f,0.f,0.f};

  float4 aReg[4], bReg[4];

  // ---- issue global loads for the K-tile starting at k0 ----
  auto load_tiles = [&](int k0) {
#pragma unroll
    for (int p = 0; p < 4; ++p) {
      const int idx = tid + p * 256;
      {
        const int r = idx >> 3, c = (idx & 7) << 2;
        aReg[p] = *(const float4*)&A[(size_t)(m0 + r) * lda + (k0 + c)];
      }
      bReg[p] = make_float4(0.f, 0.f, 0.f, 0.f);
      if (TRANSB) {
        const int n = idx >> 3, c = (idx & 7) << 2;
        if (!GUARDN || (n0 + n) < Ndim)
          bReg[p] = *(const float4*)&W[(size_t)(n0 + n) * ldw + (k0 + c)];
      } else {
        const int k = idx >> 5, n = (idx & 31) << 2;   // coalesced along n
        if (!GUARDN || (n0 + n) < Ndim)                // N always mult of 4
          bReg[p] = *(const float4*)&W[(size_t)(k0 + k) * ldw + (n0 + n)];
      }
    }
  };

  // ---- convert f32->bf16 and store staged regs into LDS buffer `buf` ----
  auto store_tiles = [&](int buf) {
    unsigned short* Asb = &As[buf][0];
    unsigned short* Bsb = &Bs[buf][0];
#pragma unroll
    for (int p = 0; p < 4; ++p) {
      const int idx = tid + p * 256;
      {
        const int r = idx >> 3, c = (idx & 7) << 2;
        uint2 pk;
        pk.x = pack2bf(aReg[p].x, aReg[p].y);
        pk.y = pack2bf(aReg[p].z, aReg[p].w);
        *(uint2*)&Asb[r * AS_STRIDE + c] = pk;
      }
      if (TRANSB) {
        const int n = idx >> 3, c = (idx & 7) << 2;
        uint2 pk;
        pk.x = pack2bf(bReg[p].x, bReg[p].y);
        pk.y = pack2bf(bReg[p].z, bReg[p].w);
        *(uint2*)&Bsb[n * BS_STRIDE + c] = pk;
      } else {
        const int k = idx >> 5, n = (idx & 31) << 2;   // scatter transpose
        Bsb[(n + 0) * BS_STRIDE + k] = f2bf(bReg[p].x);
        Bsb[(n + 1) * BS_STRIDE + k] = f2bf(bReg[p].y);
        Bsb[(n + 2) * BS_STRIDE + k] = f2bf(bReg[p].z);
        Bsb[(n + 3) * BS_STRIDE + k] = f2bf(bReg[p].w);
      }
    }
  };

  // ---- 8 WMMAs on LDS buffer `buf` ----
  auto compute = [&](int buf) {
    const unsigned short* Asb = &As[buf][0];
    const unsigned short* Bsb = &Bs[buf][0];
    // A fragments (16x32 bf16; wave32 ISA layout):
    // lanes 0-15: row M=lane, K {0..7} in v0-3, {16..23} in v4-7
    // lanes 16-31: same rows, K {8..15}, {24..31}
    v16bf af[2];
#pragma unroll
    for (int rg = 0; rg < 2; ++rg) {
      const unsigned short* ap = &Asb[(wr + rg * 16 + lr) * AS_STRIDE];
#pragma unroll
      for (int j = 0; j < 8; ++j)
        af[rg][j] = __builtin_bit_cast(__bf16, ap[hi * 8 + j]);
#pragma unroll
      for (int j = 0; j < 8; ++j)
        af[rg][8 + j] = __builtin_bit_cast(__bf16, ap[16 + hi * 8 + j]);
    }
#pragma unroll
    for (int g = 0; g < 4; ++g) {
      v16bf bf;
      const unsigned short* bp = &Bsb[(wc + g * 16 + lr) * BS_STRIDE + hi * 8];
#pragma unroll
      for (int j = 0; j < 16; ++j)
        bf[j] = __builtin_bit_cast(__bf16, bp[(j & 7) + (j >> 3) * 16]);
      acc[0][g] = __builtin_amdgcn_wmma_f32_16x16x32_bf16(false, af[0], false, bf,
                                                          (short)0, acc[0][g],
                                                          false, false);
      acc[1][g] = __builtin_amdgcn_wmma_f32_16x16x32_bf16(false, af[1], false, bf,
                                                          (short)0, acc[1][g],
                                                          false, false);
    }
  };

  // ---- double-buffered main loop: one barrier per K-step ----
  load_tiles(0);
  store_tiles(0);
  __syncthreads();
  int buf = 0;
  for (int k0 = BK; k0 < Kdim; k0 += BK) {
    load_tiles(k0);                       // global loads for tile t+1
    if (k0 + BK < Kdim) {                 // L2 prefetch for tile t+2
      __builtin_prefetch(&A[(size_t)(m0 + (tid >> 1)) * lda + k0 + BK], 0, 1);
      if (TRANSB)
        __builtin_prefetch(&W[(size_t)(n0 + (tid >> 1)) * ldw + k0 + BK], 0, 1);
      else
        __builtin_prefetch(&W[(size_t)(k0 + BK + (tid >> 3)) * ldw + n0], 0, 1);
    }
    compute(buf);                         // WMMAs on tile t (hides load latency)
    store_tiles(buf ^ 1);                 // pack tile t+1 into other buffer
    __syncthreads();
    buf ^= 1;
  }
  compute(buf);

  // ---- epilogue (C layout: VGPR i -> M = i + 8*hi, N = lane&15) ----
#pragma unroll
  for (int rg = 0; rg < 2; ++rg) {
#pragma unroll
    for (int g = 0; g < 4; ++g) {
      const int cidx = n0 + wc + g * 16 + lr;
      if (GUARDN && cidx >= Ndim) continue;
#pragma unroll
      for (int i = 0; i < 8; ++i) {
        const int r = m0 + wr + rg * 16 + hi * 8 + i;
        const size_t idx = (size_t)r * ldc + cidx;
        float v = acc[rg][g][i] * alpha;
        if (Cadd) v += Cadd[idx];
        if (epilogue == 1) v = Xin[idx] + ls[cidx] * v;
        Out[idx] = v;
      }
    }
  }
}

// ---------------------------------------------------------------------------
// LayerNorm: one block (256 threads) per row of D floats
// ---------------------------------------------------------------------------
__global__ __launch_bounds__(256)
void layernorm_kernel(const float* __restrict__ X, const float* __restrict__ s,
                      const float* __restrict__ b, float* __restrict__ Out,
                      int Ddim)
{
  __shared__ float red[256];
  const int row = blockIdx.x;
  const float* x = X + (size_t)row * Ddim;
  float* o = Out + (size_t)row * Ddim;

  float sum = 0.f;
  for (int d = threadIdx.x; d < Ddim; d += 256) sum += x[d];
  red[threadIdx.x] = sum; __syncthreads();
  for (int st = 128; st > 0; st >>= 1) {
    if (threadIdx.x < st) red[threadIdx.x] += red[threadIdx.x + st];
    __syncthreads();
  }
  const float mean = red[0] / (float)Ddim;
  __syncthreads();

  float vs = 0.f;
  for (int d = threadIdx.x; d < Ddim; d += 256) {
    float t = x[d] - mean; vs += t * t;
  }
  red[threadIdx.x] = vs; __syncthreads();
  for (int st = 128; st > 0; st >>= 1) {
    if (threadIdx.x < st) red[threadIdx.x] += red[threadIdx.x + st];
    __syncthreads();
  }
  const float inv = rsqrtf(red[0] / (float)Ddim + 1e-6f);

  for (int d = threadIdx.x; d < Ddim; d += 256)
    o[d] = (x[d] - mean) * inv * s[d] + b[d];
}

// ---------------------------------------------------------------------------
// Fused causal flash attention with ALiBi + pad mask.
// grid = (B*H, S/64), block = 64 threads; thread t handles one query row.
// K/V tiles staged with gfx1250 async global->LDS DMA (ASYNCcnt).
// QKV layout: [(b*S+s)*3D + which*D + h*64 + d]
// ---------------------------------------------------------------------------
__global__ __launch_bounds__(64)
void flash_attn_kernel(const float* __restrict__ QKV, const int* __restrict__ amask,
                       float* __restrict__ ATT)
{
  __shared__ float Kt[64][64];
  __shared__ float Vt[64][64];

  const int bh = blockIdx.x;
  const int b  = bh / Hh;
  const int h  = bh % Hh;
  const int tid = threadIdx.x;
  const int i = blockIdx.y * 64 + tid;        // query position
  const float slope = exp2f(-(0.5f + (float)h * (1.0f / 16.0f)));
  const float inv_sqrt = 0.125f;              // 1/sqrt(64)

  float q[HD], o[HD];
  const size_t qbase = ((size_t)(b * Ss + i) * 3 * Dm) + (size_t)h * HD;
#pragma unroll
  for (int d = 0; d < HD; ++d) { q[d] = QKV[qbase + d]; o[d] = 0.f; }

  float mrun = -3.0e38f, lrun = 0.f;

  // LDS byte offsets for this thread's tile rows (generic ptr low 32 bits)
  const unsigned ldsK = (unsigned)(size_t)&Kt[tid][0];
  const unsigned ldsV = (unsigned)(size_t)&Vt[tid][0];

  for (int kb = 0; kb <= blockIdx.y; ++kb) {
    // stage 64 keys + values: thread t DMAs key/value row t (16B chunks)
    {
      const size_t kvb = (size_t)(b * Ss + kb * 64 + tid) * 3 * Dm + (size_t)h * HD;
      const float* kp = QKV + kvb + Dm;
      const float* vp = QKV + kvb + 2 * Dm;
#pragma unroll
      for (int d4 = 0; d4 < HD / 4; ++d4) {
        asm volatile("global_load_async_to_lds_b128 %0, %1, off"
                     :: "v"(ldsK + d4 * 16), "v"(kp + d4 * 4) : "memory");
        asm volatile("global_load_async_to_lds_b128 %0, %1, off"
                     :: "v"(ldsV + d4 * 16), "v"(vp + d4 * 4) : "memory");
      }
      asm volatile("s_wait_asynccnt 0" ::: "memory");
    }
    __syncthreads();

    for (int j = 0; j < 64; ++j) {
      const int jj = kb * 64 + j;
      if (jj > i) break;                       // causal (-1e9 -> zero prob)
      float sc = 0.f;
#pragma unroll
      for (int d = 0; d < HD; ++d) sc += q[d] * Kt[j][d];
      sc = sc * inv_sqrt + slope * (float)(jj - i);
      if (amask[b * Ss + jj] <= 0) sc += -1e9f;

      const float newm = fmaxf(mrun, sc);
      const float scale = __expf(mrun - newm);
      const float p = __expf(sc - newm);
      lrun = lrun * scale + p;
#pragma unroll
      for (int d = 0; d < HD; ++d) o[d] = o[d] * scale + p * Vt[j][d];
      mrun = newm;
    }
    __syncthreads();
  }

  const float linv = (lrun > 0.f) ? (1.f / lrun) : 0.f;
  float* op = ATT + (size_t)(b * Ss + i) * Dm + (size_t)h * HD;
#pragma unroll
  for (int d = 0; d < HD; ++d) op[d] = o[d] * linv;
}

// ---------------------------------------------------------------------------
// SwiGLU: ACT[m,f] = silu(FF12[m,f]) * FF12[m, FF+f]
// ---------------------------------------------------------------------------
__global__ __launch_bounds__(256)
void swiglu_kernel(const float* __restrict__ FF12, float* __restrict__ ACT,
                   int total)
{
  int idx = blockIdx.x * 256 + threadIdx.x;
  if (idx >= total) return;
  int m = idx / FFm, f = idx % FFm;
  float g = FF12[(size_t)m * (2 * FFm) + f];
  float a = FF12[(size_t)m * (2 * FFm) + FFm + f];
  ACT[idx] = (g / (1.f + __expf(-g))) * a;
}

// ---------------------------------------------------------------------------
// Embedding gather: X[m,:] = tok_emb[ids[m],:]
// ---------------------------------------------------------------------------
__global__ __launch_bounds__(256)
void embed_kernel(const int* __restrict__ ids, const float* __restrict__ emb,
                  float* __restrict__ X)
{
  int m = blockIdx.x;
  const float* src = emb + (size_t)ids[m] * Dm;
  float* dst = X + (size_t)m * Dm;
  for (int d = threadIdx.x; d < Dm; d += 256) dst[d] = src[d];
}

// ---------------------------------------------------------------------------
// Host-side orchestration
// ---------------------------------------------------------------------------
static inline void launch_gemm(const float* A, const float* W, const float* Cadd,
                               const float* Xin, const float* ls, float* Out,
                               int M, int N, int K, int lda, int ldw, int ldc,
                               float alpha, int transB, int guardN, int epi,
                               hipStream_t stream)
{
  dim3 grid((N + BN - 1) / BN, (M + BM - 1) / BM);
  if (transB)
    gemm_wmma_bf16<false, true><<<grid, 256, 0, stream>>>(
        A, W, Cadd, Xin, ls, Out, N, K, lda, ldw, ldc, alpha, epi);
  else if (guardN)
    gemm_wmma_bf16<true, false><<<grid, 256, 0, stream>>>(
        A, W, Cadd, Xin, ls, Out, N, K, lda, ldw, ldc, alpha, epi);
  else
    gemm_wmma_bf16<false, false><<<grid, 256, 0, stream>>>(
        A, W, Cadd, Xin, ls, Out, N, K, lda, ldw, ldc, alpha, epi);
}

extern "C" void kernel_launch(void* const* d_in, const int* in_sizes, int n_in,
                              void* d_out, int out_size, void* d_ws, size_t ws_size,
                              hipStream_t stream)
{
  (void)in_sizes; (void)n_in; (void)out_size; (void)ws_size;

  const int*   input_ids = (const int*)  d_in[0];
  const int*   amask     = (const int*)  d_in[1];
  const float* tok_emb   = (const float*)d_in[2];
  const float* ln1_s     = (const float*)d_in[3];
  const float* ln1_b     = (const float*)d_in[4];
  const float* W_qkv     = (const float*)d_in[5];
  const float* A_qkv     = (const float*)d_in[6];
  const float* B_qkv     = (const float*)d_in[7];
  const float* W_o       = (const float*)d_in[8];
  const float* A_o       = (const float*)d_in[9];
  const float* B_o       = (const float*)d_in[10];
  const float* ls1       = (const float*)d_in[11];
  const float* ls2       = (const float*)d_in[12];
  const float* ln2_s     = (const float*)d_in[13];
  const float* ln2_b     = (const float*)d_in[14];
  const float* W_fc1     = (const float*)d_in[15];
  const float* A_fc1     = (const float*)d_in[16];
  const float* B_fc1     = (const float*)d_in[17];
  const float* W_fc2     = (const float*)d_in[18];
  const float* A_fc2     = (const float*)d_in[19];
  const float* B_fc2     = (const float*)d_in[20];
  const float* lnf_s     = (const float*)d_in[21];
  const float* lnf_b     = (const float*)d_in[22];
  float* out = (float*)d_out;

  const int M = Mrows;
  const float aR = 1.0f / (float)Rr;

  // workspace carve-up (f32)
  float* X    = (float*)d_ws;                 // M x D
  float* Hb   = X    + (size_t)M * Dm;        // M x D
  float* QKV  = Hb   + (size_t)M * Dm;        // M x 3D
  float* U    = QKV  + (size_t)M * 3 * Dm;    // M x R
  float* P    = U    + (size_t)M * Rr;        // M x 8192 (max N)
  float* FF12 = P    + (size_t)M * 2 * FFm;   // M x 8192
  float* ACT  = FF12 + (size_t)M * 2 * FFm;   // M x FF
  float* ATT  = ACT  + (size_t)M * FFm;       // M x D

  // x = tok_emb[input_ids]
  embed_kernel<<<M, 256, 0, stream>>>(input_ids, tok_emb, X);

  for (int l = 0; l < Ll; ++l) {
    // ---- attention block ----
    layernorm_kernel<<<M, 256, 0, stream>>>(X, ln1_s + (size_t)l * Dm,
                                            ln1_b + (size_t)l * Dm, Hb, Dm);
    // LoRA: U = h@A ; P = (U@B)/R ; QKV = h@W + P
    launch_gemm(Hb, A_qkv + (size_t)l * Dm * Rr, nullptr, nullptr, nullptr, U,
                M, Rr, Dm, Dm, Rr, Rr, 1.f, 0, 1, 0, stream);
    launch_gemm(U, B_qkv + (size_t)l * Rr * 3 * Dm, nullptr, nullptr, nullptr, P,
                M, 3 * Dm, Rr, Rr, 3 * Dm, 3 * Dm, aR, 0, 0, 0, stream);
    launch_gemm(Hb, W_qkv + (size_t)l * Dm * 3 * Dm, P, nullptr, nullptr, QKV,
                M, 3 * Dm, Dm, Dm, 3 * Dm, 3 * Dm, 1.f, 0, 0, 0, stream);

    flash_attn_kernel<<<dim3(Bb * Hh, Ss / 64), 64, 0, stream>>>(QKV, amask, ATT);

    // out proj + LayerScale residual: X = X + ls1 * (ATT@W_o + (ATT@A_o)@B_o/R)
    launch_gemm(ATT, A_o + (size_t)l * Dm * Rr, nullptr, nullptr, nullptr, U,
                M, Rr, Dm, Dm, Rr, Rr, 1.f, 0, 1, 0, stream);
    launch_gemm(U, B_o + (size_t)l * Rr * Dm, nullptr, nullptr, nullptr, P,
                M, Dm, Rr, Rr, Dm, Dm, aR, 0, 0, 0, stream);
    launch_gemm(ATT, W_o + (size_t)l * Dm * Dm, P, X, ls1 + (size_t)l * Dm, X,
                M, Dm, Dm, Dm, Dm, Dm, 1.f, 0, 0, 1, stream);

    // ---- MLP block ----
    layernorm_kernel<<<M, 256, 0, stream>>>(X, ln2_s + (size_t)l * Dm,
                                            ln2_b + (size_t)l * Dm, Hb, Dm);
    launch_gemm(Hb, A_fc1 + (size_t)l * Dm * Rr, nullptr, nullptr, nullptr, U,
                M, Rr, Dm, Dm, Rr, Rr, 1.f, 0, 1, 0, stream);
    launch_gemm(U, B_fc1 + (size_t)l * Rr * 2 * FFm, nullptr, nullptr, nullptr, P,
                M, 2 * FFm, Rr, Rr, 2 * FFm, 2 * FFm, aR, 0, 0, 0, stream);
    launch_gemm(Hb, W_fc1 + (size_t)l * Dm * 2 * FFm, P, nullptr, nullptr, FF12,
                M, 2 * FFm, Dm, Dm, 2 * FFm, 2 * FFm, 1.f, 0, 0, 0, stream);

    swiglu_kernel<<<(M * FFm + 255) / 256, 256, 0, stream>>>(FF12, ACT, M * FFm);

    launch_gemm(ACT, A_fc2 + (size_t)l * FFm * Rr, nullptr, nullptr, nullptr, U,
                M, Rr, FFm, FFm, Rr, Rr, 1.f, 0, 1, 0, stream);
    launch_gemm(U, B_fc2 + (size_t)l * Rr * Dm, nullptr, nullptr, nullptr, P,
                M, Dm, Rr, Rr, Dm, Dm, aR, 0, 0, 0, stream);
    launch_gemm(ACT, W_fc2 + (size_t)l * FFm * Dm, P, X, ls2 + (size_t)l * Dm, X,
                M, Dm, FFm, FFm, Dm, Dm, 1.f, 0, 0, 1, stream);
  }

  // final LN + tied LM head (W = tok_emb^T via TRANSB)
  layernorm_kernel<<<M, 256, 0, stream>>>(X, lnf_s, lnf_b, Hb, Dm);
  launch_gemm(Hb, tok_emb, nullptr, nullptr, nullptr, out,
              M, Vv, Dm, Dm, Dm, Vv, 1.f, 1, 0, 0, stream);
}
